// AttentionJacobian_34986803593380
// MI455X (gfx1250) — matrix operand order repaired and checked by reading, not compiled, not run on hardware
//
#include <hip/hip_runtime.h>
#include <hip/hip_bf16.h>

typedef __attribute__((ext_vector_type(16))) _Float16 v16h;
typedef __attribute__((ext_vector_type(8)))  float    v8f;
typedef __attribute__((ext_vector_type(4)))  unsigned int v4u;

#define BATCH   16
#define NQ      64
#define SEQ     4096
#define DH      128
#define SCALE   0.08838834764831845f   // 1/sqrt(128)
#define ST      20                     // LDS row stride in dwords (16 pairs + pad, 16B aligned)
#define NCHUNK  (SEQ / 32)

union FragH { v16h h; v4u u[2]; };

static __device__ __forceinline__ unsigned int pack2h(float x, float y) {
    _Float16 hx = (_Float16)x, hy = (_Float16)y;
    unsigned short ux = __builtin_bit_cast(unsigned short, hx);
    unsigned short uy = __builtin_bit_cast(unsigned short, hy);
    return (unsigned int)ux | ((unsigned int)uy << 16);
}

// ---------------------------------------------------------------------------
// Phase A: attention softmax.  One 256-thread WG per (b,q) row.
// attn[bq][n] = softmax_n( scale * dot(Q[bq], K[b][n]) )
// ---------------------------------------------------------------------------
__global__ __launch_bounds__(256) void attn_softmax_kernel(
    const float* __restrict__ Q, const float* __restrict__ K,
    float* __restrict__ attn)
{
    const int bq = blockIdx.x;
    const int b  = bq >> 6;
    const int t  = threadIdx.x;

    __shared__ float sQ[DH];
    __shared__ float red[256];

    if (t < DH) sQ[t] = Q[(long)bq * DH + t];
    __syncthreads();

    const float4* Kb = (const float4*)(K + (long)b * SEQ * DH);

    float s[16];
    float mx = -1e30f;
#pragma unroll
    for (int i = 0; i < 16; ++i) {
        const int n = t + i * 256;
        const float4* kr = Kb + (long)n * (DH / 4);
        float acc = 0.f;
#pragma unroll
        for (int c = 0; c < DH / 4; ++c) {
            float4 kv = kr[c];
            acc += sQ[4*c+0]*kv.x + sQ[4*c+1]*kv.y + sQ[4*c+2]*kv.z + sQ[4*c+3]*kv.w;
        }
        s[i] = acc * SCALE;
        mx = fmaxf(mx, s[i]);
    }

    // block-reduce max
    red[t] = mx; __syncthreads();
    for (int off = 128; off > 0; off >>= 1) {
        if (t < off) red[t] = fmaxf(red[t], red[t + off]);
        __syncthreads();
    }
    mx = red[0];
    __syncthreads();

    float sum = 0.f;
#pragma unroll
    for (int i = 0; i < 16; ++i) { s[i] = __expf(s[i] - mx); sum += s[i]; }

    red[t] = sum; __syncthreads();
    for (int off = 128; off > 0; off >>= 1) {
        if (t < off) red[t] = red[t] + red[t + off];
        __syncthreads();
    }
    const float inv = 1.0f / red[0];

#pragma unroll
    for (int i = 0; i < 16; ++i)
        attn[(long)bq * SEQ + t + i * 256] = s[i] * inv;
}

// ---------------------------------------------------------------------------
// Phase B: Jacobian GEMM.  One 256-thread WG (8 waves) per (b,q).
//   C = sum_n a_n * V[n] (x) K[n]   via v_wmma_f32_16x16x32_f16
//   J = scale * (C - (aV)(aK)^T)
// LDS staging layout: lds[col][pair] with pair p packing f16 rows (2p, 2p+1),
// matching the ISA 16-bit A (16x32) and B (32x16) fragment layouts so both
// fragments are two ds_load_b128 each.
// ---------------------------------------------------------------------------
__global__ __launch_bounds__(256) void attn_jacobian_kernel(
    const float* __restrict__ K, const float* __restrict__ V,
    const float* __restrict__ attn, float* __restrict__ out)
{
    const int bq = blockIdx.x;
    const int b  = bq >> 6;
    const int t  = threadIdx.x;

    __shared__ __align__(16) unsigned int ldsK[2][DH * ST];
    __shared__ __align__(16) unsigned int ldsV[2][DH * ST];
    __shared__ float sOr[256], sMr[256];
    __shared__ float sOc[DH],  sMc[DH];

    const float* Kb   = K + (long)b * SEQ * DH;
    const float* Vb   = V + (long)b * SEQ * DH;
    const float* aRow = attn + (long)bq * SEQ;

    const int c = t & 127;       // column 0..127 this thread stages
    const int g = t >> 7;        // pair-group 0/1

    float oP = 0.f, mP = 0.f;    // partial aV / aK for this column

    auto stage = [&](int n0, int buf) {
#pragma unroll
        for (int i = 0; i < 8; ++i) {
            const int p = g * 8 + i;          // pair index 0..15
            const int n = n0 + 2 * p;
            const float a0 = aRow[n];
            const float a1 = aRow[n + 1];
            const float k0 = Kb[(long)n * DH + c];
            const float k1 = Kb[(long)(n + 1) * DH + c];
            const float v0 = Vb[(long)n * DH + c];
            const float v1 = Vb[(long)(n + 1) * DH + c];
            const float av0 = a0 * v0, av1 = a1 * v1;
            ldsK[buf][c * ST + p] = pack2h(k0, k1);
            ldsV[buf][c * ST + p] = pack2h(av0, av1);
            oP += av0 + av1;
            mP += a0 * k0 + a1 * k1;
        }
    };

    // wave tiling: 8 waves in a 4x2 grid of (32-row x 64-col) blocks
    const int w      = t >> 5;
    const int lane   = t & 31;
    const int rowBlk = (w >> 1) * 32;
    const int colBlk = (w & 1) * 64;
    const int lhalf  = lane >> 4;   // 0: lanes 0-15, 1: lanes 16-31
    const int l15    = lane & 15;

    v8f acc[2][4];
#pragma unroll
    for (int mt = 0; mt < 2; ++mt)
#pragma unroll
        for (int nt = 0; nt < 4; ++nt)
            acc[mt][nt] = (v8f){0.f,0.f,0.f,0.f,0.f,0.f,0.f,0.f};

    auto compute = [&](int buf) {
        FragH A[2];
#pragma unroll
        for (int mt = 0; mt < 2; ++mt) {
            const int m    = rowBlk + mt * 16 + l15;
            const int base = m * ST + lhalf * 4;         // lanes<16: p0-3 / p8-11; lanes>=16: p4-7 / p12-15
            A[mt].u[0] = *(const v4u*)&ldsV[buf][base];
            A[mt].u[1] = *(const v4u*)&ldsV[buf][base + 8];
        }
#pragma unroll
        for (int nt = 0; nt < 4; ++nt) {
            const int cb   = colBlk + nt * 16 + l15;
            const int base = cb * ST + lhalf * 4;        // lanes<16: p0-3 / p8-11; lanes>=16: p4-7 / p12-15
            FragH Bf;
            Bf.u[0] = *(const v4u*)&ldsK[buf][base];
            Bf.u[1] = *(const v4u*)&ldsK[buf][base + 8];
#pragma unroll
            for (int mt = 0; mt < 2; ++mt)
                acc[mt][nt] = __builtin_amdgcn_wmma_f32_16x16x32_f16(
                    false, A[mt].h, false, Bf.h, (short)0, acc[mt][nt], false, false);
        }
    };

    stage(0, 0);
    __syncthreads();
#pragma unroll 1
    for (int ch = 0; ch < NCHUNK; ++ch) {
        if (ch + 1 < NCHUNK) stage((ch + 1) * 32, (ch + 1) & 1);
        compute(ch & 1);
        __syncthreads();
    }

    // reduce aV / aK vectors (term2)
    sOr[t] = oP; sMr[t] = mP;
    __syncthreads();
    if (t < DH) { sOc[t] = sOr[t] + sOr[t + 128]; sMc[t] = sMr[t] + sMr[t + 128]; }
    __syncthreads();

    // epilogue: J = scale * (C - o (x) m)
    float* outp = out + (long)bq * DH * DH;
#pragma unroll
    for (int mt = 0; mt < 2; ++mt) {
        const int rbase = rowBlk + mt * 16 + lhalf * 8;  // VGPR i -> row rbase+i
#pragma unroll
        for (int nt = 0; nt < 4; ++nt) {
            const int col  = colBlk + nt * 16 + l15;
            const float mv = sMc[col];
#pragma unroll
            for (int i = 0; i < 8; ++i) {
                const int row = rbase + i;
                outp[row * DH + col] = SCALE * (acc[mt][nt][i] - sOc[row] * mv);
            }
        }
    }
}

// ---------------------------------------------------------------------------
extern "C" void kernel_launch(void* const* d_in, const int* in_sizes, int n_in,
                              void* d_out, int out_size, void* d_ws, size_t ws_size,
                              hipStream_t stream) {
    const float* Q = (const float*)d_in[0];   // (16, 64, 128)
    const float* K = (const float*)d_in[1];   // (16, 4096, 128)
    const float* V = (const float*)d_in[2];   // (16, 4096, 128)
    float* out  = (float*)d_out;              // (16, 64, 128, 128)
    float* attn = (float*)d_ws;               // (1024, 4096) fp32 = 16.8 MB

    attn_softmax_kernel<<<BATCH * NQ, 256, 0, stream>>>(Q, K, attn);
    attn_jacobian_kernel<<<BATCH * NQ, 256, 0, stream>>>(K, V, attn, out);
}
